// Seq2Seq_48000554501045
// MI455X (gfx1250) — compile-verified, matrix operand order checked by reading, not run on hardware
//
#include <hip/hip_runtime.h>
#include <cstdint>
#include <cstddef>

// ---------------------------------------------------------------------------
// Types for CDNA5 WMMA (wave32): bf16 A/B fragments, f32 accumulator
// ---------------------------------------------------------------------------
typedef __attribute__((ext_vector_type(16))) __bf16 v16bf;
typedef __attribute__((ext_vector_type(8)))  __bf16 v8bf;
typedef __attribute__((ext_vector_type(8)))  float  v8f;

union BF16x16 { v16bf v; v8bf h8[2]; };
union F8      { v8f  v; float f[8]; };

#define B_    2048
#define H_    512
#define E_    256
#define KC_   768      // E + H, concatenated GEMM K
#define SRCL  64
#define TGTL  128
#define DECV  86
#define VPAD  96       // DECV padded to 6 x 16 WMMA tiles

#define KSTAGE   384                 // K chunk staged in LDS per pass (12 k-iters)
#define LDS_STR  (KSTAGE + 8)        // padded row stride (elements) -> conflict-free
#define BLK_M    256                 // rows per block (8 waves x 32 rows)

#if defined(__has_builtin)
#  if __has_builtin(__builtin_amdgcn_sched_barrier)
#    define SCHED_BARRIER() __builtin_amdgcn_sched_barrier(0)
#  endif
#endif
#ifndef SCHED_BARRIER
#  define SCHED_BARRIER()
#endif

__device__ __forceinline__ unsigned short f2bf(float x) {
    unsigned int u = __float_as_uint(x);
    u += 0x7FFFu + ((u >> 16) & 1u);          // round-to-nearest-even
    return (unsigned short)(u >> 16);
}
__device__ __forceinline__ float sigm(float x)      { return 1.0f / (1.0f + __expf(-x)); }
__device__ __forceinline__ float tanh_fast(float x) { return 2.0f / (1.0f + __expf(-2.0f * x)) - 1.0f; }

#define WMMA_BF16(A, Bv, C) \
    __builtin_amdgcn_wmma_f32_16x16x32_bf16(false, (A), false, (Bv), (short)0, (C), false, false)

// ---------------------------------------------------------------------------
// Fused LSTM step: gates = [emb(tok) | h_in] @ Wcat^T + bsum, then cell update.
// grid = (B/256, H/16), block = 256 (8 waves).
// Each wave: 32 rows x 16 cols x 4 gates -> 8 WMMAs / k-iter, 192 WMMAs total.
// Weights staged in LDS (shared by 8 waves). A (global) and B (LDS) fragments
// are software-pipelined with name-based ping-pong (no register copies); the
// prefetch clauses are pinned above each WMMA group with sched_barrier so the
// scheduler cannot sink them into a load-to-use stall.
// ---------------------------------------------------------------------------
__global__ __launch_bounds__(256) void lstm_step_kernel(
    const unsigned short* __restrict__ emb,   // [V, E_] bf16
    const int*            __restrict__ tokens,
    int tok_stride, int tok_off,
    const unsigned short* __restrict__ W,     // [4H, KC_] bf16  (Wih|Whh)
    const float*          __restrict__ bsum,  // [4H] = bih + bhh
    const unsigned short* __restrict__ hin,   // [B, H] bf16
    float*                __restrict__ c,     // [B, H] f32 (in-place)
    unsigned short*       __restrict__ hout)  // [B, H] bf16
{
    __shared__ unsigned short Bs[64 * LDS_STR];   // 50,176 B

    const int tid  = threadIdx.x;
    const int wave = tid >> 5;
    const int lane = tid & 31;
    const int l    = lane & 15;
    const int hx   = lane >> 4;               // half-wave
    const int m0   = blockIdx.x * BLK_M + wave * 32;
    const int j0   = blockIdx.y * 16;
    const int co   = 8 * hx;                  // per-lane K chunk offset

    // A-side: lane feeds rows m0+l (tile 0) and m0+16+l (tile 1)
    const int ar0 = m0 + l;
    const int ar1 = m0 + 16 + l;
    const int tk0 = tokens[ar0 * tok_stride + tok_off];
    const int tk1 = tokens[ar1 * tok_stride + tok_off];
    const unsigned short* aemb0 = emb + (size_t)tk0 * E_;
    const unsigned short* aemb1 = emb + (size_t)tk1 * E_;
    const unsigned short* ah0   = hin + (size_t)ar0 * H_;
    const unsigned short* ah1   = hin + (size_t)ar1 * H_;

    auto loadA = [&](int k0, v8bf* a) {       // a[0..3] = {t0c0, t0c1, t1c0, t1c1}
        const unsigned short* b0 = (k0 < E_) ? (aemb0 + k0) : (ah0 + (k0 - E_));
        const unsigned short* b1 = (k0 < E_) ? (aemb1 + k0) : (ah1 + (k0 - E_));
        a[0] = *(const v8bf*)(b0 + co);
        a[1] = *(const v8bf*)(b0 + 16 + co);
        a[2] = *(const v8bf*)(b1 + co);
        a[3] = *(const v8bf*)(b1 + 16 + co);
    };
    auto loadB = [&](int kl, BF16x16* bf) {   // 4 gate fragments from LDS
#pragma unroll
        for (int g = 0; g < 4; ++g) {
            const unsigned short* bp = &Bs[(g * 16 + l) * LDS_STR + kl + co];
            bf[g].h8[0] = *(const v8bf*)(bp);
            bf[g].h8[1] = *(const v8bf*)(bp + 16);
        }
    };

    // Cooperative stage of weight rows {g*H + j0 + 0..15 | g<4}, K in [ks, ks+384).
    // Each thread owns one row (tid>>2; 4 threads/row) and copies 12 chunks at
    // compile-time-constant offsets: no div/mod, uniform trip count, coalesced.
    const int srow = tid >> 2;                    // 0..63
    const int sc0  = (tid & 3) * 8;               // element offset within 32-elem group
    const unsigned short* ssrc =
        W + (size_t)((srow >> 4) * H_ + j0 + (srow & 15)) * KC_ + sc0;
    unsigned short* sdst = &Bs[srow * LDS_STR + sc0];
    auto stageB = [&](int ks) {
#pragma unroll
        for (int j = 0; j < KSTAGE / 32; ++j)
            *(v8bf*)(sdst + j * 32) = *(const v8bf*)(ssrc + ks + j * 32);
    };

    v8f acc[2][4] = {{{}, {}, {}, {}}, {{}, {}, {}, {}}};
    auto doWmma = [&](const v8bf* a, const BF16x16* bf) {
        BF16x16 A0, A1;
        A0.h8[0] = a[0]; A0.h8[1] = a[1];
        A1.h8[0] = a[2]; A1.h8[1] = a[3];
#pragma unroll
        for (int g = 0; g < 4; ++g) {
            acc[0][g] = WMMA_BF16(A0.v, bf[g].v, acc[0][g]);
            acc[1][g] = WMMA_BF16(A1.v, bf[g].v, acc[1][g]);
        }
    };

    v8bf    aA[4], aB[4];         // ping-pong A fragment buffers
    BF16x16 bA[4], bB[4];         // ping-pong B fragment buffers

    stageB(0);
    __syncthreads();
    loadA(0, aA);
    loadB(0, bA);

    // 24 k-iters processed in pairs; stage boundary at kk==12 (even -> pair head).
    for (int kkp = 0; kkp < KC_ / 32; kkp += 2) {
        if (kkp == 12) {
            __syncthreads();
            stageB(KSTAGE);
            __syncthreads();
            loadB(0, bA);                         // kk=12 -> kl=0 in new stage
        }
        const int k0 = kkp * 32;
        const int kl = k0 % KSTAGE;

        loadA(k0 + 32, aB);                       // A for kk+1 (always valid: kkp+1 <= 23)
        loadB(kl + 32, bB);                       // B for kk+1 (never crosses a stage)
        SCHED_BARRIER();                          // pin prefetches above this WMMA group
        doWmma(aA, bA);                           // waits: loadcnt<=4, dscnt<=8

        if (kkp + 2 < KC_ / 32) {
            loadA(k0 + 64, aA);                   // A for kk+2
            if (kkp + 2 != 12)
                loadB((kl + 64) % KSTAGE, bA);    // B for kk+2 unless stage boundary
        }
        SCHED_BARRIER();
        doWmma(aB, bB);
    }

    // Pointwise LSTM cell update (fused; gates never touch memory).
    const int   col = j0 + l;
    const float bi_ = bsum[0 * H_ + col];
    const float bf_ = bsum[1 * H_ + col];
    const float bg_ = bsum[2 * H_ + col];
    const float bo_ = bsum[3 * H_ + col];

#pragma unroll
    for (int sub = 0; sub < 2; ++sub) {
        F8 gi, gf, gg, go;
        gi.v = acc[sub][0]; gf.v = acc[sub][1]; gg.v = acc[sub][2]; go.v = acc[sub][3];
#pragma unroll
        for (int r = 0; r < 8; ++r) {
            const int    row = m0 + sub * 16 + r + 8 * hx;
            const size_t idx = (size_t)row * H_ + col;
            const float iv = sigm(gi.f[r] + bi_);
            const float fv = sigm(gf.f[r] + bf_);
            const float gv = tanh_fast(gg.f[r] + bg_);
            const float ov = sigm(go.f[r] + bo_);
            const float cn = fv * c[idx] + iv * gv;
            c[idx] = cn;
            hout[idx] = f2bf(ov * tanh_fast(cn));
        }
    }
}

// ---------------------------------------------------------------------------
// Decoder logits + argmax + token select.
// grid = B/64, block = 128; each wave: 16 rows x 96 cols (6 WMMA tiles, K=512)
// A/B software-pipelined with ping-pong register buffers.
// ---------------------------------------------------------------------------
__global__ __launch_bounds__(128) void logits_step_kernel(
    const unsigned short* __restrict__ h,     // [B, H] bf16 (this step's output)
    const unsigned short* __restrict__ fcW,   // [VPAD, H] bf16 (rows>=86 zero)
    const float*          __restrict__ fcb,   // [VPAD] (cols>=86 = -1e30)
    const int*            __restrict__ tgt,   // [B, TGTL]
    const int*            __restrict__ tmask, // [TGTL-1]
    int t,
    float*                __restrict__ out,   // [B, TGTL, DECV]
    int*                  __restrict__ tok)   // [B] next-token feedback
{
    const int tid  = threadIdx.x;
    const int wave = tid >> 5;
    const int lane = tid & 31;
    const int l    = lane & 15;
    const int hx   = lane >> 4;
    const int m0   = blockIdx.x * 64 + wave * 16;
    const int co   = 8 * hx;

    const unsigned short* ah = h + (size_t)(m0 + l) * H_;
    const unsigned short* wp[6];
#pragma unroll
    for (int tile = 0; tile < 6; ++tile)
        wp[tile] = fcW + (size_t)(tile * 16 + l) * H_;

    v8f acc[6] = {{}, {}, {}, {}, {}, {}};

    auto loadAB = [&](int k0, BF16x16& a, BF16x16* b) {
        a.h8[0] = *(const v8bf*)(ah + k0 + co);
        a.h8[1] = *(const v8bf*)(ah + k0 + 16 + co);
#pragma unroll
        for (int tile = 0; tile < 6; ++tile) {
            b[tile].h8[0] = *(const v8bf*)(wp[tile] + k0 + co);
            b[tile].h8[1] = *(const v8bf*)(wp[tile] + k0 + 16 + co);
        }
    };
    auto doWmma = [&](const BF16x16& a, const BF16x16* b) {
#pragma unroll
        for (int tile = 0; tile < 6; ++tile)
            acc[tile] = WMMA_BF16(a.v, b[tile].v, acc[tile]);
    };

    BF16x16 aA, aB, bA[6], bB[6];
    loadAB(0, aA, bA);
    for (int k0 = 0; k0 < H_; k0 += 64) {
        loadAB(k0 + 32, aB, bB);
        SCHED_BARRIER();
        doWmma(aA, bA);
        if (k0 + 64 < H_) loadAB(k0 + 64, aA, bA);
        SCHED_BARRIER();
        doWmma(aB, bB);
    }

    float bv[8]; int bx[8];
#pragma unroll
    for (int r = 0; r < 8; ++r) { bv[r] = -3.4e38f; bx[r] = 0; }

#pragma unroll
    for (int tile = 0; tile < 6; ++tile) {
        const int   col  = tile * 16 + l;
        const float bias = fcb[col];
        F8 v; v.v = acc[tile];
#pragma unroll
        for (int r = 0; r < 8; ++r) {
            const float x   = v.f[r] + bias;
            const int   row = m0 + r + 8 * hx;
            if (col < DECV)
                out[(size_t)row * (TGTL * DECV) + (size_t)(t + 1) * DECV + col] = x;
            if (x > bv[r] || (x == bv[r] && col < bx[r])) { bv[r] = x; bx[r] = col; }
        }
    }

    // Cross-lane argmax: xor masks 1..8 stay within each 16-lane half,
    // matching the C-layout (rows 0-7 in lanes 0-15, rows 8-15 in 16-31).
#pragma unroll
    for (int m = 1; m <= 8; m <<= 1) {
#pragma unroll
        for (int r = 0; r < 8; ++r) {
            const float ov = __shfl_xor(bv[r], m, 32);
            const int   oi = __shfl_xor(bx[r], m, 32);
            if (ov > bv[r] || (ov == bv[r] && oi < bx[r])) { bv[r] = ov; bx[r] = oi; }
        }
    }

    if (l == 0) {
        const int teach = tmask[t];
#pragma unroll
        for (int r = 0; r < 8; ++r) {
            const int row = m0 + r + 8 * hx;
            tok[row] = (teach > 0) ? tgt[row * TGTL + (t + 1)] : bx[r];
        }
    }
}

// ---------------------------------------------------------------------------
// Prep / init kernels
// ---------------------------------------------------------------------------
__global__ void prep_wcat(unsigned short* dst, const float* Wih, const float* Whh) {
    int i = blockIdx.x * blockDim.x + threadIdx.x;
    if (i >= 4 * H_ * KC_) return;
    int n_ = i / KC_, k = i % KC_;
    float v = (k < E_) ? Wih[n_ * E_ + k] : Whh[n_ * H_ + (k - E_)];
    dst[i] = f2bf(v);
}

__global__ void prep_bsum(float* dst, const float* bih, const float* bhh) {
    int i = blockIdx.x * blockDim.x + threadIdx.x;
    if (i < 4 * H_) dst[i] = bih[i] + bhh[i];
}

__global__ void prep_cast(unsigned short* dst, const float* src, int n) {
    int i = blockIdx.x * blockDim.x + threadIdx.x;
    if (i < n) dst[i] = f2bf(src[i]);
}

__global__ void prep_fc(unsigned short* dstW, const float* fcW, float* dstb, const float* fcb) {
    int i = blockIdx.x * blockDim.x + threadIdx.x;
    if (i < VPAD * H_) {
        int n_ = i / H_, k = i % H_;
        dstW[i] = f2bf(n_ < DECV ? fcW[n_ * H_ + k] : 0.0f);
    }
    if (i < VPAD) dstb[i] = (i < DECV) ? fcb[i] : -1.0e30f;
}

__global__ void init_state(float* c, unsigned short* h0, int* tok, const int* tgt, float* out) {
    int i = blockIdx.x * blockDim.x + threadIdx.x;
    if (i < B_ * H_) { c[i] = 0.0f; h0[i] = 0; }
    if (i < B_) tok[i] = tgt[i * TGTL];                       // x0 = tgt[:,0]
    if (i < B_ * DECV)                                        // out[:,0,:] = 0
        out[(size_t)(i / DECV) * (TGTL * DECV) + (i % DECV)] = 0.0f;
}

// ---------------------------------------------------------------------------
// Host side
// ---------------------------------------------------------------------------
extern "C" void kernel_launch(void* const* d_in, const int* in_sizes, int n_in,
                              void* d_out, int out_size, void* d_ws, size_t ws_size,
                              hipStream_t stream) {
    (void)in_sizes; (void)n_in; (void)out_size; (void)ws_size;

    const int*   src      = (const int*)d_in[0];
    const int*   tgt      = (const int*)d_in[1];
    const int*   tmask    = (const int*)d_in[2];
    const float* enc_emb  = (const float*)d_in[3];
    const float* enc_Wih  = (const float*)d_in[4];
    const float* enc_Whh  = (const float*)d_in[5];
    const float* enc_bih  = (const float*)d_in[6];
    const float* enc_bhh  = (const float*)d_in[7];
    const float* dec_emb  = (const float*)d_in[8];
    const float* dec_Wih  = (const float*)d_in[9];
    const float* dec_Whh  = (const float*)d_in[10];
    const float* dec_bih  = (const float*)d_in[11];
    const float* dec_bhh  = (const float*)d_in[12];
    const float* fc_W     = (const float*)d_in[13];
    const float* fc_b     = (const float*)d_in[14];
    float*       out      = (float*)d_out;

    char* ws = (char*)d_ws;
    size_t off = 0;
    auto alloc = [&](size_t bytes) -> void* {
        off = (off + 255) & ~(size_t)255;
        void* p = ws + off;
        off += bytes;
        return p;
    };

    unsigned short* encW  = (unsigned short*)alloc((size_t)4 * H_ * KC_ * 2);
    unsigned short* decW  = (unsigned short*)alloc((size_t)4 * H_ * KC_ * 2);
    unsigned short* eemb  = (unsigned short*)alloc((size_t)54 * E_ * 2);
    unsigned short* demb  = (unsigned short*)alloc((size_t)DECV * E_ * 2);
    unsigned short* fcWp  = (unsigned short*)alloc((size_t)VPAD * H_ * 2);
    float*          ebs   = (float*)alloc((size_t)4 * H_ * 4);
    float*          dbs   = (float*)alloc((size_t)4 * H_ * 4);
    float*          fcbp  = (float*)alloc((size_t)VPAD * 4);
    unsigned short* ha    = (unsigned short*)alloc((size_t)B_ * H_ * 2);
    unsigned short* hb2   = (unsigned short*)alloc((size_t)B_ * H_ * 2);
    float*          cbuf  = (float*)alloc((size_t)B_ * H_ * 4);
    int*            tok   = (int*)alloc((size_t)B_ * 4);

    // ---- prep (runs every call; deterministic) ----
    {
        int n = 4 * H_ * KC_;
        prep_wcat<<<(n + 255) / 256, 256, 0, stream>>>(encW, enc_Wih, enc_Whh);
        prep_wcat<<<(n + 255) / 256, 256, 0, stream>>>(decW, dec_Wih, dec_Whh);
        prep_bsum<<<(4 * H_ + 255) / 256, 256, 0, stream>>>(ebs, enc_bih, enc_bhh);
        prep_bsum<<<(4 * H_ + 255) / 256, 256, 0, stream>>>(dbs, dec_bih, dec_bhh);
        n = 54 * E_;
        prep_cast<<<(n + 255) / 256, 256, 0, stream>>>(eemb, enc_emb, n);
        n = DECV * E_;
        prep_cast<<<(n + 255) / 256, 256, 0, stream>>>(demb, dec_emb, n);
        n = VPAD * H_;
        prep_fc<<<(n + 255) / 256, 256, 0, stream>>>(fcWp, fc_W, fcbp, fc_b);
        n = B_ * H_;
        init_state<<<(n + 255) / 256, 256, 0, stream>>>(cbuf, ha, tok, tgt, out);
    }

    unsigned short* hbuf[2] = {ha, hb2};
    int cur = 0;
    const dim3 grid(B_ / BLK_M, H_ / 16);
    const dim3 block(256);

    // ---- encoder: 64 fused LSTM steps ----
    for (int t = 0; t < SRCL; ++t) {
        lstm_step_kernel<<<grid, block, 0, stream>>>(
            eemb, src, SRCL, t, encW, ebs, hbuf[cur], cbuf, hbuf[1 - cur]);
        cur ^= 1;
    }

    // ---- decoder: 127 steps of (fused LSTM, then logits+argmax+token) ----
    for (int t = 0; t < TGTL - 1; ++t) {
        lstm_step_kernel<<<grid, block, 0, stream>>>(
            demb, tok, 1, 0, decW, dbs, hbuf[cur], cbuf, hbuf[1 - cur]);
        cur ^= 1;
        logits_step_kernel<<<dim3(B_ / 64), block, 0, stream>>>(
            hbuf[cur], fcWp, fcbp, tgt, tmask, t, out, tok);
    }
}
